// DeforEncoder_84198538871272
// MI455X (gfx1250) — matrix operand order repaired and checked by reading, not compiled, not run on hardware
//
#include <hip/hip_runtime.h>
#include <hip/hip_bf16.h>

// ---------------------------------------------------------------------------
// DeforEncoder for MI455X (gfx1250, wave32, WMMA).
// All dense GEMMs run through a bf16 WMMA kernel (v_wmma_f32_16x16x32_bf16,
// fp32 accumulate). A-tiles are staged into LDS with CDNA5 async
// global->LDS loads (global_load_async_to_lds_b128, drained via
// s_wait_asynccnt). Weights are bf16-converted with zero padding to a
// multiple of the N-tile so B staging is branch-free and vectorized.
// Deformable bilinear sampling is a wave-per-(q,head) gather kernel with
// lane == channel (32 lanes == d=32).
// ---------------------------------------------------------------------------

typedef __attribute__((ext_vector_type(16))) __bf16 bf16x16;
typedef __attribute__((ext_vector_type(8)))  float  f32x8;

union FragAB { bf16x16 v; uint4 q[2]; };

__device__ __forceinline__ unsigned short f32_to_bf16(float f) {
    unsigned int u = __float_as_uint(f);
    unsigned int r = (u + 0x7FFFu + ((u >> 16) & 1u)) >> 16;   // RNE
    return (unsigned short)r;
}

// Wave-relative LDS byte offset of a __shared__ object (generic -> AS(3)).
__device__ __forceinline__ unsigned lds_off_u32(void* p) {
    return (unsigned)(unsigned long long)(__attribute__((address_space(3))) void*)p;
}

// ---------------------------------------------------------------------------
// Tiled bf16 WMMA GEMM:
//   out[M,N] = A[M,K](bf16, ld=K) * B[K,Npad](bf16, ld=ldB, zero-padded)
//              + bias (+ residual) (ReLU?) -> f32 and/or bf16
// BM=128, BN=64, BK=64; 256 threads = 8 waves; wave grid 4(M) x 2(N);
// each wave owns a 32x32 tile = 2x2 WMMA fragments; 8 WMMAs per k-tile.
// ---------------------------------------------------------------------------
#define BM 128
#define BN 64
#define BK 64

__global__ __launch_bounds__(256)
void gemm_bf16_kernel(const unsigned short* __restrict__ A,
                      const unsigned short* __restrict__ B,
                      const float* __restrict__ bias,
                      const float* __restrict__ residual,
                      float* __restrict__ outF,
                      unsigned short* __restrict__ outH,
                      int M, int N, int K, int ldB, int relu)
{
    __shared__ unsigned short lA[BM * BK];   // [m][k] row-major (16 KB)
    __shared__ unsigned short lB[BN * BK];   // [n][k] transposed stage (8 KB)

    const int tid   = threadIdx.x;
    const int lane  = tid & 31;
    const int wave  = tid >> 5;
    const int wM    = wave >> 1;       // 0..3
    const int wN    = wave & 1;        // 0..1
    const int tileM = blockIdx.y * BM;
    const int tileN = blockIdx.x * BN;

    f32x8 acc[2][2] = {};

    const int khalf = lane >> 4;       // 0 or 1
    const int lr    = lane & 15;

    for (int k0 = 0; k0 < K; k0 += BK) {
        // ---- stage A tile: thread -> (row = tid/2, 32 contiguous bf16) ----
        // CDNA5 async global->LDS DMA; INST_OFFSET applies to both addresses.
        {
            int row = tid >> 1;
            int kh  = (tid & 1) * 32;
            int gr  = tileM + row;
            unsigned short* dst = &lA[row * BK + kh];
            if (gr < M) {
                const unsigned short* src = A + (size_t)gr * K + k0 + kh;
                unsigned ldst = lds_off_u32(dst);
                asm volatile(
                    "global_load_async_to_lds_b128 %0, %1, off offset:0\n\t"
                    "global_load_async_to_lds_b128 %0, %1, off offset:16\n\t"
                    "global_load_async_to_lds_b128 %0, %1, off offset:32\n\t"
                    "global_load_async_to_lds_b128 %0, %1, off offset:48"
                    :: "v"(ldst), "v"(src) : "memory");
                if (k0 + BK < K)
                    __builtin_prefetch(src + BK, 0, 1);   // global_prefetch_b8
            } else {
                uint4 z = {0u, 0u, 0u, 0u};
                *reinterpret_cast<uint4*>(dst)      = z;
                *reinterpret_cast<uint4*>(dst + 8)  = z;
                *reinterpret_cast<uint4*>(dst + 16) = z;
                *reinterpret_cast<uint4*>(dst + 24) = z;
            }
        }
        // ---- stage B tile transposed (branch-free: B is zero-padded) ----
        {
            int k  = tid >> 2;               // 0..63
            int nb = (tid & 3) * 16;         // 0, 16, 32, 48
            const unsigned short* src = B + (size_t)(k0 + k) * ldB + tileN + nb;
            uint4 b0 = *reinterpret_cast<const uint4*>(src);
            uint4 b1 = *reinterpret_cast<const uint4*>(src + 8);
            const unsigned short* e0 = reinterpret_cast<const unsigned short*>(&b0);
            const unsigned short* e1 = reinterpret_cast<const unsigned short*>(&b1);
            #pragma unroll
            for (int j = 0; j < 8; ++j) {
                lB[(nb + j) * BK + k]     = e0[j];
                lB[(nb + 8 + j) * BK + k] = e1[j];
            }
        }
        // Drain this wave's async LDS transfers, then workgroup barrier.
        asm volatile("s_wait_asynccnt 0x0" ::: "memory");
        __syncthreads();

        #pragma unroll
        for (int ks = 0; ks < BK; ks += 32) {
            #pragma unroll
            for (int mi = 0; mi < 2; ++mi) {
                FragAB a;
                int ml = wM * 32 + mi * 16 + lr;
                // A 16x32 bf16: lanes<16 -> K{0..7,16..23}, lanes>=16 -> K{8..15,24..31}
                a.q[0] = *reinterpret_cast<const uint4*>(&lA[ml * BK + ks + khalf * 8]);
                a.q[1] = *reinterpret_cast<const uint4*>(&lA[ml * BK + ks + 16 + khalf * 8]);
                #pragma unroll
                for (int ni = 0; ni < 2; ++ni) {
                    FragAB b;
                    int nl = wN * 32 + ni * 16 + lr;
                    // B 32x16: lanes<16 -> K{0..7,16..23}, lanes>=16 -> K{8..15,24..31}
                    b.q[0] = *reinterpret_cast<const uint4*>(&lB[nl * BK + ks + khalf * 8]);
                    b.q[1] = *reinterpret_cast<const uint4*>(&lB[nl * BK + ks + 16 + khalf * 8]);
                    acc[mi][ni] = __builtin_amdgcn_wmma_f32_16x16x32_bf16(
                        false, a.v, false, b.v, (short)0, acc[mi][ni], false, false);
                }
            }
        }
        __syncthreads();
    }

    // ---- epilogue: C/D layout VGPR r -> M = r + 8*khalf, N = lane&15 ----
    #pragma unroll
    for (int mi = 0; mi < 2; ++mi) {
        #pragma unroll
        for (int ni = 0; ni < 2; ++ni) {
            #pragma unroll
            for (int r = 0; r < 8; ++r) {
                int row = tileM + wM * 32 + mi * 16 + r + 8 * khalf;
                int col = tileN + wN * 32 + ni * 16 + lr;
                if (row < M && col < N) {
                    float v = acc[mi][ni][r];
                    if (bias)     v += bias[col];
                    if (residual) v += residual[(size_t)row * N + col];
                    if (relu)     v = fmaxf(v, 0.0f);
                    if (outF) outF[(size_t)row * N + col] = v;
                    if (outH) outH[(size_t)row * N + col] = f32_to_bf16(v);
                }
            }
        }
    }
}

// ---------------------------------------------------------------------------
// Elementwise helpers
// ---------------------------------------------------------------------------
__global__ void add_cvt_kernel(const float* __restrict__ a, const float* __restrict__ b,
                               float* __restrict__ oF, unsigned short* __restrict__ oH, int n)
{
    int i = blockIdx.x * blockDim.x + threadIdx.x;
    if (i >= n) return;
    float v = a[i] + (b ? b[i] : 0.0f);
    if (oF) oF[i] = v;
    if (oH) oH[i] = f32_to_bf16(v);
}

__global__ void cvt_kernel(const float* __restrict__ in, unsigned short* __restrict__ out, int n)
{
    int i = blockIdx.x * blockDim.x + threadIdx.x;
    if (i >= n) return;
    out[i] = f32_to_bf16(in[i]);
}

// Convert weight [K,N] f32 -> [K,Npad] bf16 with zero padding (Npad mult of 64).
__global__ void cvt_pad_kernel(const float* __restrict__ in, unsigned short* __restrict__ out,
                               int K, int N, int Npad)
{
    int i = blockIdx.x * blockDim.x + threadIdx.x;
    if (i >= K * Npad) return;
    int k = i / Npad, n = i - k * Npad;
    out[i] = (n < N) ? f32_to_bf16(in[(size_t)k * N + n]) : (unsigned short)0;
}

__global__ void copy_kernel(const float* __restrict__ in, float* __restrict__ out, int n)
{
    int i = blockIdx.x * blockDim.x + threadIdx.x;
    if (i >= n) return;
    out[i] = in[i];
}

// ---------------------------------------------------------------------------
// Softmax over L*P attention logits per (query, head).  Layout [Q][8][LP].
// ---------------------------------------------------------------------------
__global__ void softmax_kernel(float* __restrict__ aw, int Q, int LP)
{
    int t = blockIdx.x * blockDim.x + threadIdx.x;
    if (t >= Q * 8) return;
    float* p = aw + (size_t)t * LP;
    float m = -1e30f;
    for (int i = 0; i < LP; ++i) m = fmaxf(m, p[i]);
    float s = 0.0f;
    for (int i = 0; i < LP; ++i) { float e = __expf(p[i] - m); p[i] = e; s += e; }
    float inv = 1.0f / s;
    for (int i = 0; i < LP; ++i) p[i] *= inv;
}

// ---------------------------------------------------------------------------
// Deformable bilinear sampling + head-wise attention aggregation.
// vproj: [L, H*W, 8, 32]   off: [Q, 8*L*4*2]   aw: [Q, 8*L*4]
// ref:   [Q, 2(agents), 2] out_bf: [Q, 256] (c = h*32 + d)
// One block per query, one wave per head, lane == channel d.
// ---------------------------------------------------------------------------
__global__ __launch_bounds__(256)
void msda_sample_kernel(const float* __restrict__ vproj,
                        const float* __restrict__ off,
                        const float* __restrict__ aw,
                        const float* __restrict__ ref,
                        unsigned short* __restrict__ outH,
                        int L, int Q, int H, int W)
{
    int q = blockIdx.x;
    if (q >= Q) return;
    int h = threadIdx.x >> 5;
    int d = threadIdx.x & 31;
    const int P = 4;
    int offStride = 8 * L * P * 2;
    int awStride  = 8 * L * P;
    int HW = H * W;
    float acc = 0.0f;

    for (int l = 0; l < L; ++l) {
        float rx = ref[(q * 2 + l) * 2 + 0];
        float ry = ref[(q * 2 + l) * 2 + 1];
        for (int p = 0; p < P; ++p) {
            int oi = q * offStride + ((h * L + l) * P + p) * 2;
            float ox = off[oi], oy = off[oi + 1];
            float wat = aw[q * awStride + (h * L + l) * P + p];
            // loc = ref + off/(W,H);  px = loc.x*W - 0.5  (align_corners=False)
            float px = rx * (float)W + ox - 0.5f;
            float py = ry * (float)H + oy - 0.5f;
            float fx = floorf(px), fy = floorf(py);
            float dx = px - fx, dy = py - fy;
            int x0 = (int)fx, y0 = (int)fy;
            float s = 0.0f;
            #pragma unroll
            for (int cy = 0; cy < 2; ++cy) {
                int yy = y0 + cy;
                bool vy = (yy >= 0) && (yy < H);
                float wy = cy ? dy : (1.0f - dy);
                #pragma unroll
                for (int cx = 0; cx < 2; ++cx) {
                    int xx = x0 + cx;
                    bool vx = (xx >= 0) && (xx < W);
                    float wx = cx ? dx : (1.0f - dx);
                    float val = 0.0f;
                    if (vx && vy) {
                        size_t idx = ((size_t)(l * HW + yy * W + xx) * 8 + h) * 32 + d;
                        val = vproj[idx];
                    }
                    s += val * wx * wy;
                }
            }
            acc += wat * s;
        }
    }
    outH[(size_t)q * 256 + h * 32 + d] = f32_to_bf16(acc);
}

// ---------------------------------------------------------------------------
// LayerNorm over C=256, one block (256 threads) per row; in-place f32 + bf16.
// ---------------------------------------------------------------------------
__global__ __launch_bounds__(256)
void layernorm_kernel(float* __restrict__ x, const float* __restrict__ g,
                      const float* __restrict__ b, unsigned short* __restrict__ outH,
                      int rows)
{
    __shared__ float red[256];
    int row = blockIdx.x;
    if (row >= rows) return;
    int t = threadIdx.x;
    float v = x[(size_t)row * 256 + t];

    red[t] = v; __syncthreads();
    for (int s = 128; s > 0; s >>= 1) { if (t < s) red[t] += red[t + s]; __syncthreads(); }
    float mean = red[0] / 256.0f;
    __syncthreads();

    float c = v - mean;
    red[t] = c * c; __syncthreads();
    for (int s = 128; s > 0; s >>= 1) { if (t < s) red[t] += red[t + s]; __syncthreads(); }
    float var = red[0] / 256.0f;

    float o = c * rsqrtf(var + 1e-5f) * g[t] + b[t];
    x[(size_t)row * 256 + t] = o;
    if (outH) outH[(size_t)row * 256 + t] = f32_to_bf16(o);
}

// ---------------------------------------------------------------------------
// Host orchestration
// ---------------------------------------------------------------------------
extern "C" void kernel_launch(void* const* d_in, const int* in_sizes, int n_in,
                              void* d_out, int out_size, void* d_ws, size_t ws_size,
                              hipStream_t stream)
{
    (void)in_sizes; (void)n_in; (void)out_size; (void)ws_size;
    const int Q = 100 * 252, Cc = 256, DFF = 1024, Hh = 100, Ww = 252;
    const int nQC = Q * Cc;

    const float* query = (const float*)d_in[0];
    const float* qpos  = (const float*)d_in[1];
    const float* value = (const float*)d_in[2];
    const float* ref   = (const float*)d_in[3];
    auto in = [&](int i) { return (const float*)d_in[i]; };

    char* ws = (char*)d_ws;
    size_t o = 0;
    auto alloc = [&](size_t bytes) -> void* {
        void* p = ws + o; o = (o + bytes + 255) & ~(size_t)255; return p;
    };
    float*          q_f   = (float*)         alloc((size_t)nQC * 4);      // current query f32
    unsigned short* q_bf  = (unsigned short*)alloc((size_t)nQC * 2);      // bf16(query)
    unsigned short* qs_bf = (unsigned short*)alloc((size_t)nQC * 2);      // bf16(query+pos)
    unsigned short* v_bf  = (unsigned short*)alloc((size_t)2 * nQC * 2);  // bf16(value)
    float*          vproj = (float*)         alloc((size_t)2 * nQC * 4);  // value projection
    unsigned short* s_bf  = (unsigned short*)alloc((size_t)nQC * 2);      // sampled (bf16)
    float*          offb  = (float*)         alloc((size_t)Q * 128 * 4);
    float*          awb   = (float*)         alloc((size_t)Q * 64 * 4);
    unsigned short* h_bf  = (unsigned short*)alloc((size_t)Q * DFF * 2);  // FFN hidden
    unsigned short* wpool = (unsigned short*)alloc((size_t)Cc * DFF * 2); // bf16 weight stage

    dim3 blk(256);
    // Convert weight (zero-padded) then run the WMMA GEMM against it.
    auto gemm_w = [&](const unsigned short* A, const float* W, const float* bias,
                      const float* resid, float* oF, unsigned short* oH,
                      int M, int N, int K, int relu) {
        int Npad = (N + 63) & ~63;
        cvt_pad_kernel<<<((K * Npad) + 255) / 256, blk, 0, stream>>>(W, wpool, K, N, Npad);
        dim3 g(Npad / BN, (M + BM - 1) / BM);
        gemm_bf16_kernel<<<g, blk, 0, stream>>>(A, wpool, bias, resid, oF, oH, M, N, K, Npad, relu);
    };

    // init: q = query (f32 + bf16), value -> bf16 (constant across blocks)
    add_cvt_kernel<<<(nQC + 255) / 256, blk, 0, stream>>>(query, nullptr, q_f, q_bf, nQC);
    cvt_kernel<<<(2 * nQC + 255) / 256, blk, 0, stream>>>(value, v_bf, 2 * nQC);

    for (int bi = 0; bi < 3; ++bi) {
        int b0 = 5 + bi * 26;
        // sa params
        const float *saWo = in(b0 + 0), *sabo = in(b0 + 1), *saWa = in(b0 + 2), *saba = in(b0 + 3),
                    *saWv = in(b0 + 4), *sabv = in(b0 + 5), *saWout = in(b0 + 6), *sabout = in(b0 + 7);
        // ca params
        const float *caWo = in(b0 + 8),  *cabo = in(b0 + 9),  *caWa = in(b0 + 10), *caba = in(b0 + 11),
                    *caWv = in(b0 + 12), *cabv = in(b0 + 13), *caWout = in(b0 + 14), *cabout = in(b0 + 15);
        const float *n1g = in(b0 + 16), *n1b = in(b0 + 17), *n2g = in(b0 + 18), *n2b = in(b0 + 19),
                    *n3g = in(b0 + 20), *n3b = in(b0 + 21);
        const float *W1 = in(b0 + 22), *b1 = in(b0 + 23), *W2 = in(b0 + 24), *b2 = in(b0 + 25);

        // ---------------- self deformable attention (L=1) ----------------
        add_cvt_kernel<<<(nQC + 255) / 256, blk, 0, stream>>>(q_f, qpos, nullptr, qs_bf, nQC);
        gemm_w(q_bf, saWv, sabv, nullptr, vproj, nullptr, Q, Cc, Cc, 0);     // v = q @ Wv + bv
        gemm_w(qs_bf, saWo, sabo, nullptr, offb, nullptr, Q, 64, Cc, 0);     // offsets
        gemm_w(qs_bf, saWa, saba, nullptr, awb, nullptr, Q, 32, Cc, 0);      // attn logits
        softmax_kernel<<<(Q * 8 + 255) / 256, blk, 0, stream>>>(awb, Q, 4);
        msda_sample_kernel<<<Q, blk, 0, stream>>>(vproj, offb, awb, ref, s_bf, 1, Q, Hh, Ww);
        gemm_w(s_bf, saWout, sabout, q_f, q_f, nullptr, Q, Cc, Cc, 0);       // out proj + residual
        layernorm_kernel<<<Q, blk, 0, stream>>>(q_f, n1g, n1b, q_bf, Q);

        // ---------------- cross deformable attention (L=2) ----------------
        add_cvt_kernel<<<(nQC + 255) / 256, blk, 0, stream>>>(q_f, qpos, nullptr, qs_bf, nQC);
        gemm_w(v_bf, caWv, cabv, nullptr, vproj, nullptr, 2 * Q, Cc, Cc, 0); // batched over agents
        gemm_w(qs_bf, caWo, cabo, nullptr, offb, nullptr, Q, 128, Cc, 0);
        gemm_w(qs_bf, caWa, caba, nullptr, awb, nullptr, Q, 64, Cc, 0);
        softmax_kernel<<<(Q * 8 + 255) / 256, blk, 0, stream>>>(awb, Q, 8);
        msda_sample_kernel<<<Q, blk, 0, stream>>>(vproj, offb, awb, ref, s_bf, 2, Q, Hh, Ww);
        gemm_w(s_bf, caWout, cabout, q_f, q_f, nullptr, Q, Cc, Cc, 0);
        layernorm_kernel<<<Q, blk, 0, stream>>>(q_f, n2g, n2b, q_bf, Q);

        // ---------------- FFN ----------------
        gemm_w(q_bf, W1, b1, nullptr, nullptr, h_bf, Q, DFF, Cc, 1);         // relu -> bf16 hidden
        gemm_w(h_bf, W2, b2, q_f, q_f, nullptr, Q, Cc, DFF, 0);              // + residual
        layernorm_kernel<<<Q, blk, 0, stream>>>(q_f, n3g, n3b, q_bf, Q);
    }

    copy_kernel<<<(nQC + 255) / 256, blk, 0, stream>>>(q_f, (float*)d_out, nQC);
}
